// GeoViG_11347303596517
// MI455X (gfx1250) — compile-verified, matrix operand order checked by reading, not run on hardware
//
#include <hip/hip_runtime.h>
#include <hip/hip_bf16.h>
#include <math.h>

// ---------------- problem constants ----------------
#define Bq   32
#define Hq   56
#define Wq   56
#define Nq   (Hq*Wq)          // 3136
#define Cq   192
#define HIDq 768
#define EPSq 1e-5f

// ---------------- tiling ----------------
#define MT        64          // tokens per block in fused kernel (3136 = 49*64)
#define MTILES    (Nq/16)     // 196 mtiles per image
#define KT_C      (Cq/32)     // 6
#define NT_C      (Cq/16)     // 12
#define NT_H      (HIDq/16)   // 48
#define KT_H      (HIDq/32)   // 24

typedef __attribute__((ext_vector_type(16))) __bf16 v16bf;
typedef __attribute__((ext_vector_type(8)))  float  v8f;

union V16 { v16bf v; uint4 q[2]; };

// Branch-free GELU (tanh form) on the native v_exp path.
// tanh(u) = 1 - 2/(1+exp(2u))  -- safe at +-inf without divergence.
__device__ __forceinline__ float gelu_f(float x) {
    float u = 0.79788456080286536f * (x + 0.044715f * x * x * x);
    float e = __expf(2.0f * u);
    float t = 1.0f - 2.0f / (e + 1.0f);
    return 0.5f * x * (1.0f + t);
}
__device__ __forceinline__ unsigned short f2bf(float x) {
    __bf16 b = (__bf16)x;
    return __builtin_bit_cast(unsigned short, b);
}
__device__ __forceinline__ void pf(const void* p) {
    __builtin_prefetch(p, 0, 1);   // -> global_prefetch_b8
}

// fragment coordinate helpers (16-bit A/B 16x32 layout, wave32):
//   k in [0,32): lane-half = (k>>3)&1 ; elem = (k&7) | ((k&16)>>1)

// ---------------------------------------------------------------------------
// Kernel 1: pack weights (f32 row-major) -> bf16 B-fragment blobs.
// Blob layout: [kt][nt][lane(32)][elem(16)] bf16  (1 KiB per 32x16 fragment).
// ---------------------------------------------------------------------------
__global__ void pack_weights(const float* __restrict__ gc_w,
                             const float* __restrict__ w1,
                             const float* __restrict__ w2,
                             unsigned short* __restrict__ gcB,
                             unsigned short* __restrict__ w1B,
                             unsigned short* __restrict__ w2B)
{
    int idx = blockIdx.x * blockDim.x + threadIdx.x;

    if (idx < KT_C * NT_C * 512) {               // gc_w: B[k][n] = gc_w[n*C + k]
        int e = idx & 15, lane = (idx >> 4) & 31;
        int nt = (idx >> 9) % NT_C, kt = idx / (512 * NT_C);
        int n = nt * 16 + (lane & 15);
        int k = kt * 32 + (e & 7) + ((e >> 3) << 4) + ((lane >> 4) << 3);
        gcB[idx] = f2bf(gc_w[n * Cq + k]);
    }
    if (idx < KT_C * NT_H * 512) {               // ffn_w1: B[k=c][n=h] = w1[h*C + c]
        int e = idx & 15, lane = (idx >> 4) & 31;
        int nt = (idx >> 9) % NT_H, kt = idx / (512 * NT_H);
        int n = nt * 16 + (lane & 15);
        int k = kt * 32 + (e & 7) + ((e >> 3) << 4) + ((lane >> 4) << 3);
        w1B[idx] = f2bf(w1[n * Cq + k]);
    }
    if (idx < KT_H * NT_C * 512) {               // ffn_w2: B[k=h][n=c] = w2[c*HID + h]
        int e = idx & 15, lane = (idx >> 4) & 31;
        int nt = (idx >> 9) % NT_C, kt = idx / (512 * NT_C);
        int n = nt * 16 + (lane & 15);
        int k = kt * 32 + (e & 7) + ((e >> 3) << 4) + ((lane >> 4) << 3);
        w2B[idx] = f2bf(w2[n * HIDq + k]);
    }
}

// ---------------------------------------------------------------------------
// Kernel 2: LN1 over C=192, one wave32 per token. f32 output to workspace.
// ---------------------------------------------------------------------------
__global__ void ln1_kernel(const float* __restrict__ x,
                           const float* __restrict__ w,
                           const float* __restrict__ bvec,
                           float* __restrict__ hbuf)
{
    int wid  = (blockIdx.x * blockDim.x + threadIdx.x) >> 5;  // token id
    int lane = threadIdx.x & 31;
    if (wid >= Bq * Nq) return;

    const float* row = x + (size_t)wid * Cq;
    float v[6];
    float s = 0.0f;
#pragma unroll
    for (int i = 0; i < 6; ++i) { v[i] = row[lane + i * 32]; s += v[i]; }
#pragma unroll
    for (int off = 16; off; off >>= 1) s += __shfl_xor(s, off, 32);
    float mean = s * (1.0f / Cq);

    float q = 0.0f;
#pragma unroll
    for (int i = 0; i < 6; ++i) { float d = v[i] - mean; q += d * d; }
#pragma unroll
    for (int off = 16; off; off >>= 1) q += __shfl_xor(q, off, 32);
    float rstd = rsqrtf(q * (1.0f / Cq) + EPSq);

    float* out = hbuf + (size_t)wid * Cq;
#pragma unroll
    for (int i = 0; i < 6; ++i) {
        int c = lane + i * 32;
        out[c] = (v[i] - mean) * rstd * w[c] + bvec[c];
    }
}

// ---------------------------------------------------------------------------
// Kernel 3: 8-neighbor stencil max minus self, bf16 A-fragment layout out.
// Every node has >=3 neighbors so the reference's -inf fixup never triggers.
// ---------------------------------------------------------------------------
__global__ void aggr_kernel(const float* __restrict__ hbuf,
                            unsigned short* __restrict__ aggrW)
{
    int wid  = (blockIdx.x * blockDim.x + threadIdx.x) >> 5;
    int lane = threadIdx.x & 31;
    if (wid >= Bq * Nq) return;

    int b = wid / Nq, tok = wid % Nq;
    int y = tok / Wq, xp = tok % Wq;
    const float* base = hbuf + (size_t)b * Nq * Cq;
    const float* hrow = base + (size_t)tok * Cq;

#pragma unroll
    for (int i = 0; i < 6; ++i) {
        int c = lane + i * 32;
        float m = -INFINITY;
#pragma unroll
        for (int dy = -1; dy <= 1; ++dy) {
#pragma unroll
            for (int dx = -1; dx <= 1; ++dx) {
                if (dy == 0 && dx == 0) continue;
                int yy = y + dy, xx = xp + dx;
                if (yy < 0 || yy >= Hq || xx < 0 || xx >= Wq) continue;
                m = fmaxf(m, base[(size_t)(yy * Wq + xx) * Cq + c]);
            }
        }
        float a  = m - hrow[c];
        int kt   = c >> 5, kk = c & 31;
        int fl   = (tok & 15) + (((kk >> 3) & 1) << 4);
        int fe   = (kk & 7) | ((kk & 16) >> 1);
        size_t fi = ((((size_t)b * MTILES + (tok >> 4)) * KT_C + kt) << 9)
                  + (size_t)fl * 16 + fe;
        aggrW[fi] = f2bf(a);
    }
}

// ---------------------------------------------------------------------------
// Kernel 4: fused  GC-GEMM -> GELU -> +res -> LN2 -> FFN1 -> GELU -> FFN2
//           -> +bias -> +res -> out.   128 threads = 4 waves, 64 tokens/block.
// All GEMMs are N-blocked by 4: one A-fragment load feeds 4 WMMAs.
// ---------------------------------------------------------------------------
__global__ void __launch_bounds__(128, 1)
fused_kernel(const float* __restrict__ x,
             const float* __restrict__ gc_bias,
             const float* __restrict__ ln2w,
             const float* __restrict__ ln2b,
             const float* __restrict__ fb1,
             const float* __restrict__ fb2,
             const unsigned short* __restrict__ aggrW,
             const unsigned short* __restrict__ gcB,
             const unsigned short* __restrict__ w1B,
             const unsigned short* __restrict__ w2B,
             float* __restrict__ out)
{
    __shared__ __align__(16) float          x1buf[MT * 193];             // 49.4 KB
    __shared__ __align__(32) unsigned short ln2f[(MT/16) * KT_C * 512];  // 24.6 KB
    __shared__ __align__(32) unsigned short ffrag[(MT/16) * KT_H * 512]; // 98.3 KB

    const int b    = blockIdx.x / (Nq / MT);
    const int t    = blockIdx.x % (Nq / MT);
    const int row0 = t * MT;
    const int tid  = threadIdx.x;
    const int w    = tid >> 5;
    const int lane = tid & 31;
    const int col  = lane & 15;
    const int lhalf = lane >> 4;
    const size_t imgoff = (size_t)b * Nq * Cq;

    // ---- phase 0: stage shortcut tile (f32) into LDS ----
    for (int i = tid; i < MT * Cq; i += blockDim.x) {
        int r = i / Cq, c = i % Cq;
        x1buf[r * 193 + c] = x[imgoff + (size_t)(row0 + r) * Cq + c];
    }
    __syncthreads();

    // ---- phase 1: GC GEMM (A = aggr frags from global, B = gcB) ----
    const int mt_g = (row0 >> 4) + w;  // this wave's global mtile
    const unsigned short* afragG =
        aggrW + ((((size_t)b * MTILES + mt_g) * KT_C) << 9);

    for (int ntb = 0; ntb < NT_C; ntb += 4) {
        v8f acc[4] = {v8f{}, v8f{}, v8f{}, v8f{}};
        for (int kt = 0; kt < KT_C; ++kt) {
            V16 a;
            const uint4* ap = (const uint4*)(afragG + ((size_t)kt << 9) + lane * 16);
            a.q[0] = ap[0]; a.q[1] = ap[1];
            if (kt + 1 < KT_C)
                pf(gcB + (((size_t)(kt + 1) * NT_C + ntb) << 9) + lane * 16);
#pragma unroll
            for (int j = 0; j < 4; ++j) {
                V16 bm;
                const uint4* bp =
                    (const uint4*)(gcB + (((size_t)kt * NT_C + ntb + j) << 9) + lane * 16);
                bm.q[0] = bp[0]; bm.q[1] = bp[1];
                acc[j] = __builtin_amdgcn_wmma_f32_16x16x32_bf16(
                    false, a.v, false, bm.v, (short)0, acc[j], false, false);
            }
        }
#pragma unroll
        for (int j = 0; j < 4; ++j) {
            int n = (ntb + j) * 16 + col;
            float bias = gc_bias[n];
#pragma unroll
            for (int r = 0; r < 8; ++r) {
                int ml = w * 16 + r + lhalf * 8;
                float g  = gelu_f(acc[j][r] + bias);
                x1buf[ml * 193 + n] = g + x1buf[ml * 193 + n];  // x1 (residual kept)
            }
        }
    }
    __syncthreads();

    // ---- phase 2: LN2 per row (2 threads/row), emit bf16 A-frags for FFN1 ----
    {
        int r    = tid >> 1;          // 128 threads -> 64 rows
        int half = tid & 1;
        int c0   = half * 96;
        float s = 0.0f;
        for (int c = c0; c < c0 + 96; ++c) s += x1buf[r * 193 + c];
        s += __shfl_xor(s, 1, 32);
        float mean = s * (1.0f / Cq);
        float q = 0.0f;
        for (int c = c0; c < c0 + 96; ++c) { float d = x1buf[r * 193 + c] - mean; q += d * d; }
        q += __shfl_xor(q, 1, 32);
        float rstd = rsqrtf(q * (1.0f / Cq) + EPSq);
        int mt = r >> 4, m15 = r & 15;
        for (int c = c0; c < c0 + 96; ++c) {
            float v = (x1buf[r * 193 + c] - mean) * rstd * ln2w[c] + ln2b[c];
            int kt = c >> 5, kk = c & 31;
            ln2f[((mt * KT_C + kt) << 9)
                 + ((m15 + (((kk >> 3) & 1) << 4)) << 4)
                 + ((kk & 7) | ((kk & 16) >> 1))] = f2bf(v);
        }
    }
    __syncthreads();

    // ---- phase 3: FFN1 (192 -> 768) + GELU, emit bf16 A-frags for FFN2 ----
    for (int ntb = 0; ntb < NT_H; ntb += 4) {
        v8f acc[4] = {v8f{}, v8f{}, v8f{}, v8f{}};
        for (int kt = 0; kt < KT_C; ++kt) {
            V16 a;
            const uint4* ap = (const uint4*)(ln2f + ((w * KT_C + kt) << 9) + lane * 16);
            a.q[0] = ap[0]; a.q[1] = ap[1];
            if (kt + 1 < KT_C)
                pf(w1B + (((size_t)(kt + 1) * NT_H + ntb) << 9) + lane * 16);
#pragma unroll
            for (int j = 0; j < 4; ++j) {
                V16 bm;
                const uint4* bp =
                    (const uint4*)(w1B + (((size_t)kt * NT_H + ntb + j) << 9) + lane * 16);
                bm.q[0] = bp[0]; bm.q[1] = bp[1];
                acc[j] = __builtin_amdgcn_wmma_f32_16x16x32_bf16(
                    false, a.v, false, bm.v, (short)0, acc[j], false, false);
            }
        }
#pragma unroll
        for (int j = 0; j < 4; ++j) {
            int h = (ntb + j) * 16 + col;
            float bias = fb1[h];
            int kt2 = h >> 5, kk = h & 31;
            int flh = (((kk >> 3) & 1) << 4);
            int fe  = (kk & 7) | ((kk & 16) >> 1);
#pragma unroll
            for (int r = 0; r < 8; ++r) {
                int m15 = r + lhalf * 8;
                float g = gelu_f(acc[j][r] + bias);
                ffrag[((w * KT_H + kt2) << 9) + ((m15 + flh) << 4) + fe] = f2bf(g);
            }
        }
    }
    __syncthreads();

    // ---- phase 4: FFN2 (768 -> 192) + bias + residual -> global out ----
    for (int ntb = 0; ntb < NT_C; ntb += 4) {
        v8f acc[4] = {v8f{}, v8f{}, v8f{}, v8f{}};
        for (int kt = 0; kt < KT_H; ++kt) {
            V16 a;
            const uint4* ap = (const uint4*)(ffrag + ((w * KT_H + kt) << 9) + lane * 16);
            a.q[0] = ap[0]; a.q[1] = ap[1];
            if (kt + 1 < KT_H)
                pf(w2B + (((size_t)(kt + 1) * NT_C + ntb) << 9) + lane * 16);
#pragma unroll
            for (int j = 0; j < 4; ++j) {
                V16 bm;
                const uint4* bp =
                    (const uint4*)(w2B + (((size_t)kt * NT_C + ntb + j) << 9) + lane * 16);
                bm.q[0] = bp[0]; bm.q[1] = bp[1];
                acc[j] = __builtin_amdgcn_wmma_f32_16x16x32_bf16(
                    false, a.v, false, bm.v, (short)0, acc[j], false, false);
            }
        }
#pragma unroll
        for (int j = 0; j < 4; ++j) {
            int n = (ntb + j) * 16 + col;
            float bias = fb2[n];
#pragma unroll
            for (int r = 0; r < 8; ++r) {
                int ml = w * 16 + r + lhalf * 8;
                float v = acc[j][r] + bias + x1buf[ml * 193 + n];
                out[imgoff + (size_t)(row0 + ml) * Cq + n] = v;
            }
        }
    }
}

// ---------------------------------------------------------------------------
extern "C" void kernel_launch(void* const* d_in, const int* in_sizes, int n_in,
                              void* d_out, int out_size, void* d_ws, size_t ws_size,
                              hipStream_t stream)
{
    const float* x      = (const float*)d_in[0];
    // d_in[1] = edge_index (unused: grid structure exploited directly)
    const float* ln1_w  = (const float*)d_in[2];
    const float* ln1_b  = (const float*)d_in[3];
    const float* gc_w   = (const float*)d_in[4];
    const float* gc_b   = (const float*)d_in[5];
    const float* ln2_w  = (const float*)d_in[6];
    const float* ln2_b  = (const float*)d_in[7];
    const float* ffn_w1 = (const float*)d_in[8];
    const float* ffn_b1 = (const float*)d_in[9];
    const float* ffn_w2 = (const float*)d_in[10];
    const float* ffn_b2 = (const float*)d_in[11];
    float* out = (float*)d_out;

    // workspace carve-up (256B aligned)
    char* ws = (char*)d_ws;
    size_t off = 0;
    auto alloc = [&](size_t bytes) {
        char* p = ws + off;
        off += (bytes + 255) & ~(size_t)255;
        return p;
    };
    float*          hbuf  = (float*)         alloc((size_t)Bq * Nq * Cq * 4); // 77.1 MB
    unsigned short* aggrW = (unsigned short*)alloc((size_t)Bq * Nq * Cq * 2); // 38.5 MB
    unsigned short* gcB   = (unsigned short*)alloc((size_t)KT_C * NT_C * 512 * 2);
    unsigned short* w1B   = (unsigned short*)alloc((size_t)KT_C * NT_H * 512 * 2);
    unsigned short* w2B   = (unsigned short*)alloc((size_t)KT_H * NT_C * 512 * 2);

    // 1) weight repack to WMMA B-fragment order (bf16)
    {
        int total = KT_C * NT_H * 512;  // largest of the three (147456)
        pack_weights<<<(total + 255) / 256, 256, 0, stream>>>(
            gc_w, ffn_w1, ffn_w2, gcB, w1B, w2B);
    }
    // 2) LN1 (one wave per token)
    {
        long long waves = (long long)Bq * Nq;
        long long thr   = waves * 32;
        ln1_kernel<<<(int)((thr + 255) / 256), 256, 0, stream>>>(
            x, ln1_w, ln1_b, hbuf);
    }
    // 3) neighbor max - self -> bf16 A-fragments
    {
        long long waves = (long long)Bq * Nq;
        long long thr   = waves * 32;
        aggr_kernel<<<(int)((thr + 255) / 256), 256, 0, stream>>>(hbuf, aggrW);
    }
    // 4) fused WMMA chain
    {
        int blocks = Bq * (Nq / MT);   // 32 * 49 = 1568
        fused_kernel<<<blocks, 128, 0, stream>>>(
            x, gc_b, ln2_w, ln2_b, ffn_b1, ffn_b2,
            aggrW, gcB, w1B, w2B, out);
    }
}
